// MultiHeadAttention_75685913690640
// MI455X (gfx1250) — matrix operand order repaired and checked by reading, not compiled, not run on hardware
//
#include <hip/hip_runtime.h>
#include <hip/hip_bf16.h>
#include <math.h>

#define BB   4
#define HH   16
#define SSZ  2048
#define DQK  64
#define DVV  64
#define HIDD 1024

typedef __attribute__((ext_vector_type(16))) __bf16 v16bf;
typedef __attribute__((ext_vector_type(8)))  __bf16 v8bf;
typedef __attribute__((ext_vector_type(8)))  float  v8f;

static __device__ __forceinline__ __bf16 bf(float x) { return (__bf16)x; }

// ---------------------------------------------------------------------------
// Converter: f32 -> bf16, elementwise, 8 elems/thread (b128 in, b128 out)
// ---------------------------------------------------------------------------
__global__ void cvt_bf16_kernel(const float* __restrict__ src,
                                __bf16* __restrict__ dst, int n8)
{
  int i = blockIdx.x * blockDim.x + threadIdx.x;
  if (i >= n8) return;
  const float* s = src + (size_t)i * 8;
  v8bf o;
  #pragma unroll
  for (int t = 0; t < 8; ++t) o[t] = bf(s[t]);
  *(v8bf*)(dst + (size_t)i * 8) = o;
}

// ---------------------------------------------------------------------------
// Converter: V [BH][S][Dv] f32 -> Vt [BH][Dv][S] bf16 (LDS-tiled transpose)
// grid: (S/32, BH), block: 256
// ---------------------------------------------------------------------------
__global__ void transpose_v_kernel(const float* __restrict__ vsrc,
                                   __bf16* __restrict__ vt)
{
  __shared__ float tile[32][DVV + 1];
  const int bh = blockIdx.y;
  const int s0 = blockIdx.x * 32;
  const float* V = vsrc + (size_t)bh * SSZ * DVV;
  __bf16*     T  = vt   + (size_t)bh * DVV * SSZ;

  #pragma unroll
  for (int it = 0; it < 8; ++it) {            // 32 rows x 64 cols, 256 thr
    int idx = it * 256 + threadIdx.x;
    int i = idx >> 6, d = idx & 63;
    tile[i][d] = V[(size_t)(s0 + i) * DVV + d];
  }
  __syncthreads();
  #pragma unroll
  for (int it = 0; it < 8; ++it) {
    int idx = it * 256 + threadIdx.x;
    int d = idx >> 5, j = idx & 31;           // contiguous s -> coalesced
    T[(size_t)d * SSZ + s0 + j] = bf(tile[j][d]);
  }
}

// ---------------------------------------------------------------------------
// Kernel 1: per (b,h, 16-query-tile) wave: scores -> softmax (per-lane online
// stats, one final lane-merge) -> write attn -> ctx = p @ V (ctx bf16,
// stored as [B, S, H*Dv])
// ---------------------------------------------------------------------------
__global__ __launch_bounds__(32) void mha_attn_kernel(
    const __bf16* __restrict__ qb, const __bf16* __restrict__ kb_,
    const __bf16* __restrict__ vtb, const float* __restrict__ maskg,
    float* __restrict__ attng, __bf16* __restrict__ ctxg)
{
  __shared__ __align__(32) __bf16 pshm[16][32];  // D->A transpose staging

  const int lane = threadIdx.x & 31;
  const int half = lane >> 4;
  const int lmod = lane & 15;

  const int qt = blockIdx.x % (SSZ / 16);
  const int bh = blockIdx.x / (SSZ / 16);
  const int b  = bh / HH;
  const int h  = bh % HH;
  const int qbase = qt * 16;

  const __bf16* Q  = qb  + (size_t)bh * SSZ * DQK;
  const __bf16* K  = kb_ + (size_t)bh * SSZ * DQK;
  const __bf16* Vt = vtb + (size_t)bh * DVV * SSZ;
  const float*  M  = maskg + (size_t)bh * SSZ * SSZ;
  float* A = attng + (size_t)bh * SSZ * SSZ;
  __bf16* C = ctxg + (size_t)b * SSZ * HIDD + (size_t)h * DVV;

  // Per-lane view of D layout: lane l, reg r -> row (qbase+r+8*half), col lmod
  const float* Mbase = M + (size_t)(qbase + 8 * half) * SSZ + lmod;
  float*       Abase = A + (size_t)(qbase + 8 * half) * SSZ + lmod;

  // ---- Q tile in WMMA A layout (two 16x32 bf16 matrices), pure vec loads ---
  v16bf a0, a1;
  {
    const __bf16* qrow = Q + (size_t)(qbase + lmod) * DQK;
    v8bf l0 = *(const v8bf*)(qrow +  0 + 16 * half);
    v8bf h0 = *(const v8bf*)(qrow +  8 + 16 * half);
    v8bf l1 = *(const v8bf*)(qrow + 32 + 16 * half);
    v8bf h1 = *(const v8bf*)(qrow + 40 + 16 * half);
    #pragma unroll
    for (int t = 0; t < 8; ++t) {
      a0[t] = l0[t]; a0[8 + t] = h0[t];
      a1[t] = l1[t]; a1[8 + t] = h1[t];
    }
  }

  // B operand, contiguous k order (same as A): lane l: n=l%16; elem e -> k=16*(l/16)+e
  // -> 16 contiguous bf16 = one 32-byte vector load
  auto qkTile = [&](int keybase) {
    const __bf16* kr = K + (size_t)(keybase + lmod) * DQK + 16 * half;
    v16bf b0 = *(const v16bf*)(kr);
    v16bf b1 = *(const v16bf*)(kr + 32);
    v8f c = {};
    c = __builtin_amdgcn_wmma_f32_16x16x32_bf16(false, a0, false, b0,
                                                (short)0, c, false, false);
    c = __builtin_amdgcn_wmma_f32_16x16x32_bf16(false, a1, false, b1,
                                                (short)0, c, false, false);
    return c;
  };

  // ---- pass 1: per-lane online (max, sum-of-exp); 1 exp per element.
  // Unroll x2: two independent WMMA chains per iteration so the scheduler can
  // fill the WMMA->VALU hazard slots (4 co-exec NOPs otherwise).
  float lm[8], ls[8];
  #pragma unroll
  for (int r = 0; r < 8; ++r) { lm[r] = -1e30f; ls[r] = 0.f; }

  #pragma unroll 2
  for (int kt = 0; kt < SSZ / 16; ++kt) {
    v8f s = qkTile(kt * 16);
    #pragma unroll
    for (int r = 0; r < 8; ++r) {
      float val = s[r] + Mbase[(size_t)r * SSZ + kt * 16];
      float mn  = fmaxf(lm[r], val);
      float mi  = fminf(lm[r], val);
      float e   = __expf(mi - mn);
      bool  nw  = val > lm[r];            // new max arrived
      ls[r] = fmaf(ls[r], nw ? e : 1.0f, nw ? 1.0f : e);
      lm[r] = mn;
    }
  }

  // ---- single lane-merge of (m, s) across the 16 columns each half owns ----
  float rm[8], rinv[8];
  #pragma unroll
  for (int r = 0; r < 8; ++r) {
    float m = lm[r], s = ls[r];
    #pragma unroll
    for (int off = 1; off < 16; off <<= 1) {
      float om = __shfl_xor(m, off, 16);
      float os = __shfl_xor(s, off, 16);
      float mn = fmaxf(m, om);
      s = s * __expf(m - mn) + os * __expf(om - mn);
      m = mn;
    }
    rm[r]   = m;
    rinv[r] = 1.0f / s;
  }

  // ---- pass 2: recompute, normalize, store attn, accumulate ctx = p @ V ----
  v8f acc[4];
  #pragma unroll
  for (int t = 0; t < 4; ++t) acc[t] = (v8f){};

  for (int c = 0; c < SSZ / 32; ++c) {
    const int kb = c * 32;
    v8f s0 = qkTile(kb);
    v8f s1 = qkTile(kb + 16);
    #pragma unroll
    for (int r = 0; r < 8; ++r) {
      float p0 = __expf(s0[r] + Mbase[(size_t)r * SSZ + kb]      - rm[r]) * rinv[r];
      float p1 = __expf(s1[r] + Mbase[(size_t)r * SSZ + kb + 16] - rm[r]) * rinv[r];
      Abase[(size_t)r * SSZ + kb]      = p0;   // coalesced over 16 lanes
      Abase[(size_t)r * SSZ + kb + 16] = p1;
      pshm[r + 8 * half][lmod]      = bf(p0);  // stage for D->A transpose
      pshm[r + 8 * half][16 + lmod] = bf(p1);
    }
    __syncthreads();

    v16bf pa;  // p tile re-read in A layout (16x32 bf16)
    {
      v8bf lo = *(const v8bf*)(&pshm[lmod][ 0 + 16 * half]);
      v8bf hi = *(const v8bf*)(&pshm[lmod][ 8 + 16 * half]);
      #pragma unroll
      for (int t = 0; t < 8; ++t) { pa[t] = lo[t]; pa[8 + t] = hi[t]; }
    }

    #pragma unroll
    for (int n0 = 0; n0 < 4; ++n0) {
      // B layout from transposed V: (k,n) = Vt[n0*16+n][kb+k] -> 32B vec load
      v16bf bv = *(const v16bf*)(Vt + (size_t)(n0 * 16 + lmod) * SSZ
                                    + kb + 16 * half);
      acc[n0] = __builtin_amdgcn_wmma_f32_16x16x32_bf16(false, pa, false, bv,
                                                        (short)0, acc[n0],
                                                        false, false);
    }
    __syncthreads();
  }

  // ---- store ctx (bf16) in [B, S, H*Dv] order ----
  #pragma unroll
  for (int n0 = 0; n0 < 4; ++n0) {
    #pragma unroll
    for (int r = 0; r < 8; ++r)
      C[(size_t)(qbase + r + 8 * half) * HIDD + n0 * 16 + lmod]
          = bf(acc[n0][r]);
  }
}

// ---------------------------------------------------------------------------
// Kernel 2: projection  out[8192,1024] = x[8192,1024] @ W[1024,1024]^T
// (x, W in bf16; all-vector loads)
// ---------------------------------------------------------------------------
__global__ __launch_bounds__(32) void mha_proj_kernel(
    const __bf16* __restrict__ xg,  // [B*S, HID] bf16 (ctx, reordered)
    const __bf16* __restrict__ wg,  // [HID, HID] bf16 (W_proj)
    float* __restrict__ outg)       // [B*S, HID] f32
{
  const int lane = threadIdx.x & 31;
  const int half = lane >> 4;
  const int lmod = lane & 15;
  const int rowbase = blockIdx.x * 16;
  const int obase   = blockIdx.y * 16;

  const __bf16* xr = xg + (size_t)(rowbase + lmod) * HIDD;
  const __bf16* wr = wg + (size_t)(obase   + lmod) * HIDD;

  v8f acc = {};
  #pragma unroll 2
  for (int c = 0; c < HIDD / 32; ++c) {
    const int kb = c * 32;
    v8bf lo = *(const v8bf*)(xr + kb +  0 + 16 * half);
    v8bf hi = *(const v8bf*)(xr + kb +  8 + 16 * half);
    v16bf am;
    #pragma unroll
    for (int t = 0; t < 8; ++t) { am[t] = lo[t]; am[8 + t] = hi[t]; }
    v16bf bm = *(const v16bf*)(wr + kb + 16 * half);
    acc = __builtin_amdgcn_wmma_f32_16x16x32_bf16(false, am, false, bm,
                                                  (short)0, acc, false, false);
  }
  #pragma unroll
  for (int r = 0; r < 8; ++r)
    outg[(size_t)(rowbase + r + 8 * half) * HIDD + obase + lmod] = acc[r];
}

// ---------------------------------------------------------------------------
extern "C" void kernel_launch(void* const* d_in, const int* in_sizes, int n_in,
                              void* d_out, int out_size, void* d_ws, size_t ws_size,
                              hipStream_t stream) {
  (void)in_sizes; (void)n_in; (void)out_size; (void)ws_size;
  const float* q    = (const float*)d_in[0];
  const float* k    = (const float*)d_in[1];
  const float* v    = (const float*)d_in[2];
  const float* mask = (const float*)d_in[3];
  const float* W    = (const float*)d_in[4];

  float* out  = (float*)d_out;                        // [B, S, HID]
  float* attn = out + (size_t)BB * SSZ * HIDD;        // [B, H, S, S]

  // workspace carve-up (all 256B aligned)
  const size_t nQK = (size_t)BB * HH * SSZ * DQK;     // 8.4M elems
  char* wsp = (char*)d_ws;
  __bf16* qbf  = (__bf16*)wsp;                wsp += nQK * 2;              // 16MB
  __bf16* kbf  = (__bf16*)wsp;                wsp += nQK * 2;              // 16MB
  __bf16* vtbf = (__bf16*)wsp;                wsp += nQK * 2;              // 16MB
  __bf16* wbf  = (__bf16*)wsp;                wsp += (size_t)HIDD * HIDD * 2; // 2MB
  __bf16* ctx  = (__bf16*)wsp;                                            // 16MB

  // pre-convert inputs to bf16 (and transpose V)
  cvt_bf16_kernel<<<(int)(nQK / 8 / 256), 256, 0, stream>>>(q, qbf, (int)(nQK / 8));
  cvt_bf16_kernel<<<(int)(nQK / 8 / 256), 256, 0, stream>>>(k, kbf, (int)(nQK / 8));
  cvt_bf16_kernel<<<(HIDD * HIDD / 8 / 256), 256, 0, stream>>>(
      W, wbf, HIDD * HIDD / 8);
  {
    dim3 gt(SSZ / 32, BB * HH);
    transpose_v_kernel<<<gt, 256, 0, stream>>>(v, vtbf);
  }

  mha_attn_kernel<<<BB * HH * (SSZ / 16), 32, 0, stream>>>(qbf, kbf, vtbf,
                                                           mask, attn, ctx);
  dim3 g2(BB * SSZ / 16, HIDD / 16);
  mha_proj_kernel<<<g2, 32, 0, stream>>>(ctx, wbf, out);
}
